// VectorQuantization_34239479284121
// MI455X (gfx1250) — matrix operand order repaired and checked by reading, not compiled, not run on hardware
//
#include <hip/hip_runtime.h>

#define NROWS  32768
#define DDIM   256
#define KCODES 4096
#define BSTRIDE 272   // LDS row stride (ushort elems) to stagger banks

typedef __attribute__((ext_vector_type(16))) __bf16 v16bf;
typedef __attribute__((ext_vector_type(8)))  float  v8f;

#if __has_builtin(__builtin_amdgcn_global_load_async_to_lds_b128) && \
    __has_builtin(__builtin_amdgcn_s_wait_asynccnt)
#define HAVE_ASYNC_LDS 1
typedef int v4i __attribute__((ext_vector_type(4)));
typedef __attribute__((address_space(1))) v4i* gv4i;   // non-const: builtin param
typedef __attribute__((address_space(3))) v4i* lv4i;
#else
#define HAVE_ASYNC_LDS 0
#endif

__device__ __forceinline__ unsigned short f2bf(float f) {
  unsigned int u = __float_as_uint(f);
  unsigned int r = u + 0x7FFFu + ((u >> 16) & 1u);  // round-to-nearest-even
  return (unsigned short)(r >> 16);
}
__device__ __forceinline__ float bf2f(unsigned short b) {
  return __uint_as_float(((unsigned int)b) << 16);
}

// ---------------------------------------------------------------------------
// Kernel 1: embed [D][K] f32  ->  embedT hi/lo [K][D] bf16, plus ||e_k||^2
// ---------------------------------------------------------------------------
__global__ __launch_bounds__(256) void vq_prep(const float* __restrict__ embed,
    unsigned short* __restrict__ eT_hi, unsigned short* __restrict__ eT_lo,
    float* __restrict__ e_norm) {
  int k = blockIdx.x * 256 + threadIdx.x;   // coalesced reads across k
  float nrm = 0.f;
  for (int d = 0; d < DDIM; ++d) {
    float f = embed[d * KCODES + k];
    nrm += f * f;
    unsigned short h = f2bf(f);
    unsigned short l = f2bf(f - bf2f(h));
    eT_hi[k * DDIM + d] = h;
    eT_lo[k * DDIM + d] = l;
  }
  e_norm[k] = nrm;
}

// ---------------------------------------------------------------------------
// Kernel 2: split-bf16 WMMA GEMM (x . E) fused with argmin over K.
// Block = 256 threads = 8 waves; wave owns 16 rows; 128 rows per block.
// A fragments register-resident (ISA 16-bit A 16x32 layout); B tiles streamed
// into double-buffered LDS with GLOBAL_LOAD_ASYNC_TO_LDS_B128 (ASYNCcnt),
// overlapping the DMA for tile c+1 with the 24 WMMAs of tile c.
// ---------------------------------------------------------------------------
__global__ __launch_bounds__(256) void vq_gemm_argmin(
    const float* __restrict__ x,
    const unsigned short* __restrict__ eT_hi,
    const unsigned short* __restrict__ eT_lo,
    const float* __restrict__ e_norm,
    int* __restrict__ idx_i, float* __restrict__ idx_f) {
  __shared__ unsigned short ldsB[2][2][16 * BSTRIDE];  // [buf][hi/lo][col*BSTRIDE+d]

  const int tid  = threadIdx.x;
  const int wave = tid >> 5;
  const int lane = tid & 31;
  const int half = lane >> 4;   // 0: lanes 0-15, 1: lanes 16-31
  const int lm   = lane & 15;
  const int rowBase = blockIdx.x * 128 + wave * 16;

  // per-thread staging coordinates: 32B contiguous chunk per array
  const int colS  = tid >> 4;         // 0..15
  const int dOffS = (tid & 15) * 16;  // 0..240

  auto stage = [&](int c, int buf) {
    const size_t g = (size_t)(c * 16 + colS) * DDIM + dOffS;
    const int l = colS * BSTRIDE + dOffS;
#if HAVE_ASYNC_LDS
    // IOFFSET applies to both global and LDS addresses (ISA 10.x async formula)
    __builtin_amdgcn_global_load_async_to_lds_b128(
        (gv4i)(v4i*)(eT_hi + g), (lv4i)(v4i*)&ldsB[buf][0][l], 0, 0);
    __builtin_amdgcn_global_load_async_to_lds_b128(
        (gv4i)(v4i*)(eT_hi + g), (lv4i)(v4i*)&ldsB[buf][0][l], 16, 0);
    __builtin_amdgcn_global_load_async_to_lds_b128(
        (gv4i)(v4i*)(eT_lo + g), (lv4i)(v4i*)&ldsB[buf][1][l], 0, 0);
    __builtin_amdgcn_global_load_async_to_lds_b128(
        (gv4i)(v4i*)(eT_lo + g), (lv4i)(v4i*)&ldsB[buf][1][l], 16, 0);
#else
    const uint4* gh = (const uint4*)(eT_hi + g);
    const uint4* gl = (const uint4*)(eT_lo + g);
    uint4* sh = (uint4*)&ldsB[buf][0][l];
    uint4* sl = (uint4*)&ldsB[buf][1][l];
    sh[0] = gh[0]; sh[1] = gh[1];
    sl[0] = gl[0]; sl[1] = gl[1];
#endif
  };

  // --- A fragments: lane holds row (rowBase+lm).
  // 16-bit A 16x32 layout: lanes 0-15 -> K {0..7, 16..23}; lanes 16-31 -> +8.
  v16bf a_hi[8], a_lo[8];
  {
    const float* xr = x + (size_t)(rowBase + lm) * DDIM;
    #pragma unroll
    for (int ds = 0; ds < 8; ++ds) {
      const int d0 = ds * 32 + half * 8;
      float4 c0 = *(const float4*)(xr + d0);
      float4 c1 = *(const float4*)(xr + d0 + 4);
      float4 c2 = *(const float4*)(xr + d0 + 16);
      float4 c3 = *(const float4*)(xr + d0 + 20);
      float fv[16] = {c0.x,c0.y,c0.z,c0.w, c1.x,c1.y,c1.z,c1.w,
                      c2.x,c2.y,c2.z,c2.w, c3.x,c3.y,c3.z,c3.w};
      union { v16bf v; unsigned short u[16]; } Ah, Al;
      #pragma unroll
      for (int j = 0; j < 16; ++j) {
        unsigned short h = f2bf(fv[j]);
        Ah.u[j] = h;
        Al.u[j] = f2bf(fv[j] - bf2f(h));
      }
      a_hi[ds] = Ah.v;
      a_lo[ds] = Al.v;
    }
  }

  float minv[8];
  int   mini[8];
  #pragma unroll
  for (int v = 0; v < 8; ++v) { minv[v] = 3.4e38f; mini[v] = 0; }

  stage(0, 0);   // prime the pipeline

  for (int c = 0; c < KCODES / 16; ++c) {
    const int buf = c & 1;
#if HAVE_ASYNC_LDS
    __builtin_amdgcn_s_wait_asynccnt(0);   // my DMA into ldsB[buf] landed
#endif
    __syncthreads();                       // everyone's landed; prev buf free
    if (c + 1 < KCODES / 16) stage(c + 1, buf ^ 1);   // overlap with WMMAs

    v8f acc = {0.f,0.f,0.f,0.f,0.f,0.f,0.f,0.f};
    #pragma unroll
    for (int ds = 0; ds < 8; ++ds) {
      // 16-bit B 32x16 layout: lane col = lm; lanes 0-15 K {0..7,16..23}, 16-31 +8
      const int kb = ds * 32 + half * 8;
      union { v16bf v; uint4 q[2]; } BH, BL;
      BH.q[0] = *(const uint4*)&ldsB[buf][0][lm * BSTRIDE + kb];
      BH.q[1] = *(const uint4*)&ldsB[buf][0][lm * BSTRIDE + kb + 16];
      BL.q[0] = *(const uint4*)&ldsB[buf][1][lm * BSTRIDE + kb];
      BL.q[1] = *(const uint4*)&ldsB[buf][1][lm * BSTRIDE + kb + 16];
      v16bf bh = BH.v;
      v16bf bl = BL.v;
      acc = __builtin_amdgcn_wmma_f32_16x16x32_bf16(false, a_hi[ds], false, bh,
                                                    (short)0, acc, false, false);
      acc = __builtin_amdgcn_wmma_f32_16x16x32_bf16(false, a_hi[ds], false, bl,
                                                    (short)0, acc, false, false);
      acc = __builtin_amdgcn_wmma_f32_16x16x32_bf16(false, a_lo[ds], false, bh,
                                                    (short)0, acc, false, false);
    }

    const int colIdx = c * 16 + lm;        // C layout: lane's N = lane&15
    const float en = e_norm[colIdx];       // one L2-resident cacheline / tile
    #pragma unroll
    for (int v = 0; v < 8; ++v) {          // VGPR v -> row v + 8*half
      float s = en - 2.0f * acc[v];        // ||x||^2 omitted (row-constant)
      if (s < minv[v]) { minv[v] = s; mini[v] = colIdx; }  // first-min ties
    }
  }

  // Cross-lane argmin within each 16-lane group (all hold same row per VGPR)
  #pragma unroll
  for (int v = 0; v < 8; ++v) {
    float mv = minv[v]; int mi = mini[v];
    #pragma unroll
    for (int off = 8; off >= 1; off >>= 1) {
      float ov = __shfl_xor(mv, off, 32);
      int   oi = __shfl_xor(mi, off, 32);
      if (ov < mv || (ov == mv && oi < mi)) { mv = ov; mi = oi; }
    }
    minv[v] = mv; mini[v] = mi;
  }
  if (lm == 0) {   // lane 0 -> rows 0..7, lane 16 -> rows 8..15
    #pragma unroll
    for (int v = 0; v < 8; ++v) {
      int row = rowBase + half * 8 + v;
      idx_i[row] = mini[v];
      idx_f[row] = (float)mini[v];
    }
  }
}

// ---------------------------------------------------------------------------
// Kernel 3: histogram of code usage (integer-valued float adds -> exact)
// ---------------------------------------------------------------------------
__global__ __launch_bounds__(256) void vq_count(const int* __restrict__ idx,
                                                float* __restrict__ counts) {
  int n = blockIdx.x * 256 + threadIdx.x;
  atomicAdd(&counts[idx[n]], 1.0f);
}

// ---------------------------------------------------------------------------
// Kernel 4: cluster_size EMA + deterministic sum (single block)
// ---------------------------------------------------------------------------
__global__ __launch_bounds__(1024) void vq_cluster(const float* __restrict__ cs_old,
    const float* __restrict__ counts, float* __restrict__ cs_new,
    float* __restrict__ nsum) {
  __shared__ float red[1024];
  int tid = threadIdx.x;
  float local = 0.f;
  for (int k = tid; k < KCODES; k += 1024) {
    float v = cs_old[k] * 0.99f + counts[k] * (1.0f - 1e-5f);
    cs_new[k] = v;
    local += v;
  }
  red[tid] = local;
  __syncthreads();
  for (int s = 512; s > 0; s >>= 1) {
    if (tid < s) red[tid] += red[tid + s];
    __syncthreads();
  }
  if (tid == 0) nsum[0] = red[0];
}

// ---------------------------------------------------------------------------
// Kernel 5: embed_avg EMA, Laplace-normalized embed_new (+ transposed copy)
// ---------------------------------------------------------------------------
__global__ __launch_bounds__(256) void vq_embed(const float* __restrict__ embed_avg,
    const float* __restrict__ cs_new, const float* __restrict__ nsum,
    float* __restrict__ ea_out, float* __restrict__ emb_out,
    float* __restrict__ embT_new) {
  int id = blockIdx.x * 256 + threadIdx.x;   // over D*K, layout d*K+k
  int k = id & (KCODES - 1);
  int d = id >> 12;
  float n  = nsum[0];
  float ea = embed_avg[id] * 0.99f;
  ea_out[id] = ea;
  float cs = (cs_new[k] + 1e-5f) / (n + KCODES * 1e-5f) * n;
  float e  = ea / cs;
  emb_out[id] = e;
  embT_new[k * DDIM + d] = e;
}

// ---------------------------------------------------------------------------
// Kernel 6: gather quantize (straight-through == quantize) + MSE partials
// ---------------------------------------------------------------------------
__global__ __launch_bounds__(256) void vq_quant(const float* __restrict__ x,
    const float* __restrict__ embT_new, const int* __restrict__ idx,
    float* __restrict__ q_out, float* __restrict__ dpart) {
  __shared__ float red[256];
  int n = blockIdx.x, d = threadIdx.x;
  int k = idx[n];
  float q  = embT_new[k * DDIM + d];   // coalesced via transposed table
  float xv = x[(size_t)n * DDIM + d];
  q_out[(size_t)n * DDIM + d] = q;
  float df = q - xv;
  red[d] = df * df;
  __syncthreads();
  for (int s = 128; s > 0; s >>= 1) {
    if (d < s) red[d] += red[d + s];
    __syncthreads();
  }
  if (d == 0) dpart[n] = red[0];
}

// ---------------------------------------------------------------------------
// Kernel 7: deterministic final reduction for diff
// ---------------------------------------------------------------------------
__global__ __launch_bounds__(1024) void vq_diff(const float* __restrict__ dpart,
                                                float* __restrict__ diff) {
  __shared__ float red[1024];
  int tid = threadIdx.x;
  float local = 0.f;
  for (int i = tid; i < NROWS; i += 1024) local += dpart[i];
  red[tid] = local;
  __syncthreads();
  for (int s = 512; s > 0; s >>= 1) {
    if (tid < s) red[tid] += red[tid + s];
    __syncthreads();
  }
  if (tid == 0) diff[0] = red[0] / 8388608.0f;
}

// ---------------------------------------------------------------------------
extern "C" void kernel_launch(void* const* d_in, const int* in_sizes, int n_in,
                              void* d_out, int out_size, void* d_ws, size_t ws_size,
                              hipStream_t stream) {
  const float* x            = (const float*)d_in[0];   // [32,1024,256]
  const float* embed        = (const float*)d_in[1];   // [256,4096]
  const float* cluster_size = (const float*)d_in[2];   // [4096]
  const float* embed_avg    = (const float*)d_in[3];   // [256,4096]
  float* out = (float*)d_out;

  // workspace layout (bytes)
  char* ws = (char*)d_ws;
  unsigned short* eT_hi = (unsigned short*)(ws + 0);        // 2 MB
  unsigned short* eT_lo = (unsigned short*)(ws + 2097152);  // 2 MB
  float* e_norm   = (float*)(ws + 4194304);                 // 16 KB
  int*   idx_i    = (int*)  (ws + 4210688);                 // 128 KB
  float* counts   = (float*)(ws + 4341760);                 // 16 KB
  float* nsum     = (float*)(ws + 4358144);                 // 4 B (+pad)
  float* embT_new = (float*)(ws + 4358400);                 // 4 MB
  float* dpart    = (float*)(ws + 8552704);                 // 128 KB

  // output tuple, concatenated flat in return order
  float* o_q    = out;              // 8388608
  float* o_idx  = out + 8388608;    // 32768
  float* o_diff = out + 8421376;    // 1
  float* o_cs   = out + 8421377;    // 4096
  float* o_ea   = out + 8425473;    // 1048576
  float* o_emb  = out + 9474049;    // 1048576

  vq_prep<<<KCODES / 256, 256, 0, stream>>>(embed, eT_hi, eT_lo, e_norm);
  vq_gemm_argmin<<<NROWS / 128, 256, 0, stream>>>(x, eT_hi, eT_lo, e_norm,
                                                  idx_i, o_idx);
  (void)hipMemsetAsync(counts, 0, KCODES * sizeof(float), stream);
  vq_count<<<NROWS / 256, 256, 0, stream>>>(idx_i, counts);
  vq_cluster<<<1, 1024, 0, stream>>>(cluster_size, counts, o_cs, nsum);
  vq_embed<<<(DDIM * KCODES) / 256, 256, 0, stream>>>(embed_avg, o_cs, nsum,
                                                      o_ea, o_emb, embT_new);
  vq_quant<<<NROWS, 256, 0, stream>>>(x, embT_new, idx_i, o_q, dpart);
  vq_diff<<<1, 1024, 0, stream>>>(dpart, o_diff);
}
